// FpnCov1dFSCls_13795434955110
// MI455X (gfx1250) — matrix operand order repaired and checked by reading, not compiled, not run on hardware
//
#include <hip/hip_runtime.h>
#include <hip/hip_bf16.h>

// ---------------------------------------------------------------------------
// FpnCov1dFSCls for MI455X (gfx1250, wave32, WMMA).
//
// Roofline: expert GEMM = 32 x [4096x256]@[256x1000] = 67 GFLOP (bf16 WMMA,
// f32 acc). Packed bf16 weights = 16.5MB -> L2 resident. Weight fragments are
// staged into LDS (double-buffered 2x64KB, async global->LDS engine when the
// toolchain exposes it) and shared by all 8 waves of a block; per-block weight
// traffic is 2.1MB -> ~540MB total, far below the WMMA compute time. fs
// softmax weights are applied to per-rule accumulators (8 FMA / 8 WMMAs) so
// the hot loop is ~1 VALU op per WMMA. Gating fc1 (8 GFLOP) also on WMMA.
// ---------------------------------------------------------------------------

#define RR 32
#define BB 4096
#define FF 256
#define NCLS 1000
#define ATT 248
#define HID 124
#define NCT 63          // ceil(1000/16) column tiles for expert GEMM

typedef __attribute__((ext_vector_type(16))) __bf16 v16bf;
typedef __attribute__((ext_vector_type(8)))  float  v8f;
typedef int v4i_ls __attribute__((vector_size(4 * sizeof(int))));

// Async global->LDS path (gfx1250 ASYNCcnt engine), guarded so we fall back
// to the VGPR-bounce copy if this toolchain lacks the builtin.
#if defined(__HIP_DEVICE_COMPILE__) && \
    __has_builtin(__builtin_amdgcn_global_load_async_to_lds_b128)
#define HAS_ASYNC_LDS 1
#else
#define HAS_ASYNC_LDS 0
#endif

__device__ __forceinline__ void wait_async_zero() {
#if defined(__HIP_DEVICE_COMPILE__)
#if __has_builtin(__builtin_amdgcn_s_wait_asynccnt)
    __builtin_amdgcn_s_wait_asynccnt(0);
#elif HAS_ASYNC_LDS
    asm volatile("s_wait_asynccnt 0x0" ::: "memory");
#endif
#endif
}

// Workspace layout (bytes)
#define WP_ELEMS   (32LL * 8 * NCT * 32 * 16)     // 8,257,536 bf16 (consq_w packed)
#define FP_ELEMS   (8 * 8 * 32 * 16)              // 32,768 bf16 (fc1_w packed)
#define OFF_WP     0
#define OFF_FP     (WP_ELEMS * 2)                 // 16,515,072
#define OFF_FSINI  (OFF_FP + FP_ELEMS * 2)        // + 65,536
#define OFF_FS     (OFF_FSINI + (RR * BB) * 4)    // + 524,288

// A-fragment K index for 16-bit WMMA (from ISA layout table):
//   lanes 0-15 : j<8 -> K=j,     j>=8 -> K=16+(j-8)
//   lanes 16-31: j<8 -> K=8+j,   j>=8 -> K=24+(j-8)
__device__ __forceinline__ int a_kidx(int lane, int j) {
    return ((j >> 3) << 4) + ((lane >> 4) << 3) + (j & 7);
}

// ---------------------------------------------------------------------------
// Kernel 0: pack consq_w and fc1_w (fp32) into bf16 WMMA B-fragment layout.
// B fragment uses the same K order as A (a_kidx): lanes 0-15 hold K=0..7,16..23,
// lanes 16-31 hold K=8..15,24..31; N = lane&15.
// ---------------------------------------------------------------------------
__global__ __launch_bounds__(256) void pack_kernel(const float* __restrict__ consq_w,
                                                   const float* __restrict__ fc1_w,
                                                   __bf16* __restrict__ wp,
                                                   __bf16* __restrict__ fp) {
    const long long NW = WP_ELEMS;
    const long long total = NW + FP_ELEMS;
    for (long long i = (long long)blockIdx.x * blockDim.x + threadIdx.x;
         i < total; i += (long long)gridDim.x * blockDim.x) {
        if (i < NW) {
            int j    = (int)(i & 15);
            int lane = (int)((i >> 4) & 31);
            long long rest = i >> 9;
            int ct = (int)(rest % NCT);
            long long r2 = rest / NCT;
            int kc = (int)(r2 & 7);
            int r  = (int)(r2 >> 3);
            int K  = kc * 32 + a_kidx(lane, j);        // < 256 always
            int c  = ct * 16 + (lane & 15);
            float v = (c < NCLS) ? consq_w[(long long)(r * FF + K) * NCLS + c] : 0.f;
            wp[i] = (__bf16)v;
        } else {
            int t    = (int)(i - NW);
            int j    = t & 15;
            int lane = (t >> 4) & 31;
            int ct   = (t >> 9) & 7;
            int kc   = (t >> 12) & 7;
            int K    = kc * 32 + a_kidx(lane, j);
            int col  = ct * 16 + (lane & 15);
            float v = (K < ATT && col < HID) ? fc1_w[K * HID + col] : 0.f;
            fp[t] = (__bf16)v;
        }
    }
}

// ---------------------------------------------------------------------------
// Kernel 1: gating network. 1 block (256 thr = 8 waves) per 16 samples.
// Conv stages on VALU (wave handles 2 samples), fc1 via bf16 WMMA
// (wave w owns output-column tile ct=w), fc2 via shuffle + LDS atomics.
// ---------------------------------------------------------------------------
#define XSTR 260            // padded diff buffer (258 used)
#define H1STR 132           // per-channel padded h1 (129 used)
__global__ __launch_bounds__(256) void gating_kernel(
        const float* __restrict__ data, const float* __restrict__ proto,
        const float* __restrict__ conv1_w, const float* __restrict__ conv1_b,
        const float* __restrict__ conv2_w, const float* __restrict__ conv2_b,
        const __bf16* __restrict__ fpack,
        const float* __restrict__ fc1_b, const float* __restrict__ fc2_w,
        const float* __restrict__ fc2_b, float* __restrict__ fs_ini) {
    __shared__ float s_x[16 * XSTR];          // diff (stage 1-2), then h3 (stage 3+)
    __shared__ float s_h1[16 * 4 * H1STR];    // padded pooled conv1 output
    __shared__ float s_fsacc[16];

    const int wave = threadIdx.x >> 5;
    const int lane = threadIdx.x & 31;

    if (threadIdx.x < 16) s_fsacc[threadIdx.x] = 0.f;

    // ---- stage 1: diff = data[b] - proto[r], padded by 1 on each side ----
    for (int it = 0; it < 2; ++it) {
        int s  = wave * 2 + it;
        int gs = blockIdx.x * 16 + s;          // gs = r*B + b
        int r  = gs >> 12;
        int b  = gs & (BB - 1);
        float* xp = s_x + s * XSTR;
#pragma unroll
        for (int j = 0; j < 8; ++j) {
            int f = lane * 8 + j;
            xp[1 + f] = data[b * FF + f] - proto[r * FF + f];
        }
        if (lane == 0) { xp[0] = 0.f; xp[257] = 0.f; xp[258] = 0.f; xp[259] = 0.f; }
    }
    __syncthreads();

    // uniform conv weights (scalar-load friendly)
    float w1[4][5], cb1[4], w2[4][4][5], cb2[4];
#pragma unroll
    for (int c = 0; c < 4; ++c) {
        cb1[c] = conv1_b[c]; cb2[c] = conv2_b[c];
#pragma unroll
        for (int k = 0; k < 5; ++k) w1[c][k] = conv1_w[c * 5 + k];
#pragma unroll
        for (int ci = 0; ci < 4; ++ci)
#pragma unroll
            for (int k = 0; k < 5; ++k) w2[c][ci][k] = conv2_w[(c * 4 + ci) * 5 + k];
    }

    // ---- stage 2: conv1 + relu + avgpool2 -> h1 (4 x 127, padded) ----
    for (int it = 0; it < 2; ++it) {
        int s = wave * 2 + it;
        const float* xp = s_x + s * XSTR;
        float* h1 = s_h1 + s * 4 * H1STR;
        if (lane < 4) { h1[lane * H1STR] = 0.f; h1[lane * H1STR + 128] = 0.f; }
#pragma unroll
        for (int t = 0; t < 4; ++t) {
            int p = lane + 32 * t;
            if (p < 127) {
                float win[6];
#pragma unroll
                for (int j = 0; j < 6; ++j) win[j] = xp[2 * p + j];
#pragma unroll
                for (int c = 0; c < 4; ++c) {
                    float ca = cb1[c], cb = cb1[c];
#pragma unroll
                    for (int k = 0; k < 5; ++k) {
                        ca += w1[c][k] * win[k];
                        cb += w1[c][k] * win[k + 1];
                    }
                    h1[c * H1STR + 1 + p] = 0.5f * (fmaxf(ca, 0.f) + fmaxf(cb, 0.f));
                }
            }
        }
    }
    __syncthreads();

    // ---- stage 3: conv2 + relu + avgpool2 -> h3 (flattened c*62+p), over s_x ----
    for (int it = 0; it < 2; ++it) {
        int s = wave * 2 + it;
        const float* h1 = s_h1 + s * 4 * H1STR;
        float* h3 = s_x + s * XSTR;             // reuse diff buffer (248 <= 260)
#pragma unroll
        for (int t = 0; t < 2; ++t) {
            int p2 = lane + 32 * t;
            if (p2 < 62) {
                float win[4][6];
#pragma unroll
                for (int ci = 0; ci < 4; ++ci)
#pragma unroll
                    for (int j = 0; j < 6; ++j) win[ci][j] = h1[ci * H1STR + 2 * p2 + j];
#pragma unroll
                for (int c = 0; c < 4; ++c) {
                    float ca = cb2[c], cb = cb2[c];
#pragma unroll
                    for (int ci = 0; ci < 4; ++ci)
#pragma unroll
                        for (int k = 0; k < 5; ++k) {
                            ca += w2[c][ci][k] * win[ci][k];
                            cb += w2[c][ci][k] * win[ci][k + 1];
                        }
                    h3[c * 62 + p2] = 0.5f * (fmaxf(ca, 0.f) + fmaxf(cb, 0.f));
                }
            }
        }
    }
    __syncthreads();

    // ---- stage 4: fc1 via WMMA, 16 samples x (248 -> 124); wave owns ct=wave ----
    v8f acc = {};
    const float* h3m = s_x + (lane & 15) * XSTR;     // A row M = lane&15 = sample
#pragma unroll
    for (int kc = 0; kc < 8; ++kc) {
        v16bf a;
#pragma unroll
        for (int j = 0; j < 16; ++j) {
            int K = kc * 32 + a_kidx(lane, j);
            float v = (K < ATT) ? h3m[K] : 0.f;
            a[j] = (__bf16)v;
        }
        const v16bf bfrag = *(const v16bf*)(fpack + ((kc * 8 + wave) * 32 + lane) * 16);
        acc = __builtin_amdgcn_wmma_f32_16x16x32_bf16(false, a, false, bfrag,
                                                      (short)0, acc, false, false);
    }

    // ---- fc2: sum_j relu(h+b1)*w2 ; D row M = v + 8*(lane>=16), col = ct*16+(lane&15)
    int jcol = wave * 16 + (lane & 15);
    float w2f = (jcol < HID) ? fc2_w[jcol] : 0.f;
    float b1f = (jcol < HID) ? fc1_b[jcol] : 0.f;
    float rsum[8];
#pragma unroll
    for (int v = 0; v < 8; ++v) {
        float p = fmaxf(acc[v] + b1f, 0.f) * w2f;
        p += __shfl_xor(p, 1, 16);
        p += __shfl_xor(p, 2, 16);
        p += __shfl_xor(p, 4, 16);
        p += __shfl_xor(p, 8, 16);
        rsum[v] = p;
    }
    if ((lane & 15) == 0) {
        int rowbase = (lane >> 4) << 3;     // lane0 -> rows 0..7, lane16 -> rows 8..15
#pragma unroll
        for (int v = 0; v < 8; ++v) atomicAdd(&s_fsacc[rowbase + v], rsum[v]);
    }
    __syncthreads();
    if (threadIdx.x < 16) {
        int gs = blockIdx.x * 16 + threadIdx.x;
        fs_ini[gs] = tanhf(s_fsacc[threadIdx.x] + fc2_b[0]);
    }
}

// ---------------------------------------------------------------------------
// Kernel 2: softmax over rules. One wave per batch row, lane = rule.
// ---------------------------------------------------------------------------
__global__ __launch_bounds__(256) void softmax_kernel(const float* __restrict__ fs_ini,
                                                      float* __restrict__ fs) {
    int b = blockIdx.x * 8 + (threadIdx.x >> 5);
    int lane = threadIdx.x & 31;
    float v = fs_ini[lane * BB + b];           // fs_ini[r*B + b]
    float m = v;
#pragma unroll
    for (int off = 16; off >= 1; off >>= 1) m = fmaxf(m, __shfl_xor(m, off, 32));
    float e = __expf(v - m);
    float s = e;
#pragma unroll
    for (int off = 16; off >= 1; off >>= 1) s += __shfl_xor(s, off, 32);
    fs[b * RR + lane] = e / s;
}

// ---------------------------------------------------------------------------
// Kernel 3: fused expert GEMM with LDS-staged weights.
//   out[b,c] = sum_r fs[b,r]*( sum_f data[b,f]*W[r,f,c] + consq_b[r,c] )
// Block = 8 waves x 16 rows = 128 batch rows; block owns 8 column tiles.
// Weight fragments for (rule-group of 8) x (8 k-chunks) x (1 ct) = 64KB are
// staged into a double-buffered LDS pipeline (async global->LDS engine when
// available) and consumed by all waves. Unscaled bf16 A-fragments live in
// registers; fs is applied per rule with 8 FMAs after an 8-WMMA chain.
// ---------------------------------------------------------------------------
#define STG_ELEMS (8 * 8 * 32 * 16)        // 32768 bf16 = 64KB per buffer
__global__ __launch_bounds__(256) void expert_kernel(const float* __restrict__ data,
                                                     const __bf16* __restrict__ wp,
                                                     const float* __restrict__ fs,
                                                     const float* __restrict__ consq_b,
                                                     float* __restrict__ out) {
    __shared__ __bf16 s_stage[2][STG_ELEMS];   // 2 x 64KB staged B fragments
    __shared__ float  s_fs[128 * 33];          // fs for the block's 128 rows

    const int tid  = threadIdx.x;
    const int wave = tid >> 5;
    const int lane = tid & 31;
    const int bg     = blockIdx.x >> 3;        // 0..31 row groups of 128
    const int ctbase = (blockIdx.x & 7) * 8;   // 8 column tiles per block
    const int b0     = bg * 128;

    // ---- fs tile -> LDS (128 rows x 32 rules, stride 33) ----
#pragma unroll
    for (int i = 0; i < 16; ++i) {
        int idx = tid + 256 * i;               // 4096 values
        int m = idx >> 5, r = idx & 31;
        s_fs[m * 33 + r] = fs[(b0 + m) * RR + r];
    }

    // ---- unscaled bf16 A fragments (this wave's 16 rows, all 8 k-chunks) ----
    const int brow = b0 + wave * 16 + (lane & 15);
    const int koff = (lane & 16) >> 1;         // 0 or 8 (A-layout K sub-block)
    v16bf afrag[8];
#pragma unroll
    for (int kc = 0; kc < 8; ++kc) {
        const float* dp = data + brow * FF + kc * 32 + koff;
        const float4 q0 = *(const float4*)(dp);
        const float4 q1 = *(const float4*)(dp + 4);
        const float4 q2 = *(const float4*)(dp + 16);
        const float4 q3 = *(const float4*)(dp + 20);
        float d[16] = {q0.x, q0.y, q0.z, q0.w, q1.x, q1.y, q1.z, q1.w,
                       q2.x, q2.y, q2.z, q2.w, q3.x, q3.y, q3.z, q3.w};
#pragma unroll
        for (int j = 0; j < 16; ++j) afrag[kc][j] = (__bf16)d[j];
    }

    // ---- LDS fill: stage s = (ctoff<<2) | rgroup ; 64KB = 4096 x 16B ----
    auto fill = [&](int s) {
        int ct = ctbase + (s >> 2);
        if (ct < NCT) {
            int r0 = (s & 3) * 8;
            __bf16* dst = s_stage[s & 1];
#pragma unroll
            for (int i = 0; i < 16; ++i) {
                int g   = tid + 256 * i;           // granule of 8 bf16 (16B)
                int blk = g >> 6;                  // 0..63 = rr*8 + kc
                int e   = g & 63;
                int rr  = blk >> 3, kc = blk & 7;
                long long src = (((long long)((r0 + rr) * 8 + kc)) * NCT + ct) * 512
                                + (e << 3);
                int dstoff = (blk << 9) + (e << 3);
#if HAS_ASYNC_LDS
                __builtin_amdgcn_global_load_async_to_lds_b128(
                    (__attribute__((address_space(1))) v4i_ls*)(&wp[src]),
                    (__attribute__((address_space(3))) v4i_ls*)(&dst[dstoff]),
                    0, 0);
#else
                *(uint4*)(&dst[dstoff]) = *(const uint4*)(&wp[src]);
#endif
            }
        }
    };

    fill(0);
    const int hb = lane >> 4;
    const int n  = lane & 15;
    const float* fsw = s_fs + (wave * 16 + 8 * hb) * 33;
    v8f accf = {};

    for (int s = 0; s < 32; ++s) {
        wait_async_zero();                      // async fills done before barrier
        __syncthreads();                        // stage (s) ready; buf (s+1)&1 free
        if (s + 1 < 32) fill(s + 1);            // overlap next fill with compute
        int ct = ctbase + (s >> 2);
        if (ct < NCT) {                         // uniform: EXEC all-ones for WMMA
            const __bf16* stg = s_stage[s & 1];
            int r0 = (s & 3) * 8;
#pragma unroll
            for (int rr = 0; rr < 8; ++rr) {
                v8f accr = {};
#pragma unroll
                for (int kc = 0; kc < 8; ++kc) {
                    const v16bf bfrag =
                        *(const v16bf*)(stg + (((rr << 3) + kc) << 9) + (lane << 4));
                    accr = __builtin_amdgcn_wmma_f32_16x16x32_bf16(
                        false, afrag[kc], false, bfrag, (short)0, accr, false, false);
                }
                int r = r0 + rr;
#pragma unroll
                for (int v = 0; v < 8; ++v)
                    accf[v] += fsw[v * 33 + r] * accr[v];
            }
        }
        if ((s & 3) == 3) {                     // this ct done: bias + store
            if (ct < NCT) {
                int col = ct * 16 + n;
                if (col < NCLS) {
                    float bias[8] = {0.f, 0.f, 0.f, 0.f, 0.f, 0.f, 0.f, 0.f};
                    for (int r = 0; r < 32; ++r) {
                        float cb = consq_b[r * NCLS + col];
#pragma unroll
                        for (int v = 0; v < 8; ++v) bias[v] += fsw[v * 33 + r] * cb;
                    }
#pragma unroll
                    for (int v = 0; v < 8; ++v)
                        out[(long long)(b0 + wave * 16 + 8 * hb + v) * NCLS + col] =
                            accf[v] + bias[v];
                }
            }
            accf = v8f{};
        }
    }
}

// ---------------------------------------------------------------------------
extern "C" void kernel_launch(void* const* d_in, const int* in_sizes, int n_in,
                              void* d_out, int out_size, void* d_ws, size_t ws_size,
                              hipStream_t stream) {
    (void)in_sizes; (void)n_in; (void)out_size; (void)ws_size;
    const float* data    = (const float*)d_in[0];
    const float* proto   = (const float*)d_in[1];
    const float* conv1_w = (const float*)d_in[2];
    const float* conv1_b = (const float*)d_in[3];
    const float* conv2_w = (const float*)d_in[4];
    const float* conv2_b = (const float*)d_in[5];
    const float* fc1_w   = (const float*)d_in[6];
    const float* fc1_b   = (const float*)d_in[7];
    const float* fc2_w   = (const float*)d_in[8];
    const float* fc2_b   = (const float*)d_in[9];
    const float* consq_w = (const float*)d_in[10];
    const float* consq_b = (const float*)d_in[11];

    char* ws = (char*)d_ws;
    __bf16* wp     = (__bf16*)(ws + OFF_WP);
    __bf16* fp     = (__bf16*)(ws + OFF_FP);
    float*  fs_ini = (float*)(ws + OFF_FSINI);
    float*  fs     = (float*)(ws + OFF_FS);
    float*  out    = (float*)d_out;

    pack_kernel<<<4096, 256, 0, stream>>>(consq_w, fc1_w, wp, fp);
    gating_kernel<<<(RR * BB) / 16, 256, 0, stream>>>(data, proto, conv1_w, conv1_b,
                                                      conv2_w, conv2_b, fp, fc1_b,
                                                      fc2_w, fc2_b, fs_ini);
    softmax_kernel<<<BB / 8, 256, 0, stream>>>(fs_ini, fs);
    expert_kernel<<<BB / 16 / 8, 256, 0, stream>>>(data, wp, fs, consq_b, out);
}
